// DecoderLSTM_B_72662256713743
// MI455X (gfx1250) — matrix operand order repaired and checked by reading, not compiled, not run on hardware
//
#include <hip/hip_runtime.h>
#include <hip/hip_bf16.h>

typedef __attribute__((ext_vector_type(16))) _Float16 v16h;
typedef __attribute__((ext_vector_type(8)))  _Float16 v8h;
typedef __attribute__((ext_vector_type(8)))  float    v8f;

// ---------------------------------------------------------------------------
// WMMA helpers (gfx1250, wave32). D = A(16x32 f16) x B(32x16 f16) + C(16x16 f32)
// A layout: lane l<16 -> row l, K = k0..k0+15 contiguous;
//           lane l>=16 -> row l-16, K = k0+16..k0+31 contiguous
// B layout: lane l<16 -> col l, K = k0..k0+15 contiguous;
//           lane l>=16 -> col l-16, K = k0+16..k0+31 contiguous
// C/D: lane l<16 vgpr r -> (M=r, N=l); lane l>=16 vgpr r -> (M=8+r, N=l-16)
// ---------------------------------------------------------------------------
__device__ __forceinline__ v16h load_a16(const _Float16* rowp, int k0, int hi) {
  union { v16h v; struct { v8h l; v8h h; } s; } u;
  u.s.l = *(const v8h*)(rowp + k0 + hi * 16);
  u.s.h = *(const v8h*)(rowp + k0 + hi * 16 + 8);
  return u.v;
}

__device__ __forceinline__ v16h load_b16(const _Float16* base, int ld, int n0,
                                         int k0, int l16, int hi) {
  return *(const v16h*)(base + (size_t)(n0 + l16) * ld + k0 + hi * 16);
}

__device__ __forceinline__ v8f wmma_f16(v16h a, v16h b, v8f c) {
  return __builtin_amdgcn_wmma_f32_16x16x32_f16(false, a, false, b,
                                                (short)0, c, false, false);
}

__device__ __forceinline__ float sigmoid_(float x) {
  return 1.0f / (1.0f + __expf(-x));
}
__device__ __forceinline__ float tanh_(float x) {
  float e = __expf(-2.0f * fabsf(x));
  float t = (1.0f - e) / (1.0f + e);
  return x >= 0.0f ? t : -t;
}

// ---------------------------------------------------------------------------
// Kernel 0: f32 -> f16 conversions + hbias[b][n] = b_ih+b_hh+h0@W_hh.T (const)
// ---------------------------------------------------------------------------
__global__ void prep_kernel(const float* __restrict__ emb, const float* __restrict__ wih,
                            const float* __restrict__ whh, const float* __restrict__ bih,
                            const float* __restrict__ bhh, const float* __restrict__ h0,
                            const float* __restrict__ wpred,
                            _Float16* __restrict__ emb16, _Float16* __restrict__ wpred16,
                            _Float16* __restrict__ wihx16, _Float16* __restrict__ wh16,
                            float* __restrict__ hbias) {
  int tid = blockIdx.x * blockDim.x + threadIdx.x;
  int stride = gridDim.x * blockDim.x;
  const int NV = 32000 * 128;
  for (int i = tid; i < NV; i += stride) {
    emb16[i] = (_Float16)emb[i];
    wpred16[i] = (_Float16)wpred[i];
  }
  for (int i = tid; i < 512 * 128; i += stride) {
    int n = i >> 7, k = i & 127;
    wihx16[i] = (_Float16)wih[n * 256 + k];        // x-part cols [0,128)
    wh16[i]   = (_Float16)wih[n * 256 + 128 + k];  // h-part cols [128,256)
  }
  for (int i = tid; i < 32 * 512; i += stride) {
    int b = i >> 9, n = i & 511;
    float acc = bih[n] + bhh[n];
    const float* hp = h0 + b * 128;
    const float* wp = whh + n * 128;
#pragma unroll 8
    for (int k = 0; k < 128; ++k) acc += hp[k] * wp[k];
    hbias[i] = acc;
  }
}

// ---------------------------------------------------------------------------
// Kernel 1: xw[row][n] = emb16[seq[row]] @ W_x.T + hbias[row/256][n]
// rows = B*S = 8192, N = 4H = 512, K = E = 128. One 16-row tile per block.
// ---------------------------------------------------------------------------
__global__ void __launch_bounds__(256) xw_kernel(const int* __restrict__ seq,
                                                 const _Float16* __restrict__ emb16,
                                                 const _Float16* __restrict__ wihx16,
                                                 const float* __restrict__ hbias,
                                                 float* __restrict__ xw) {
  int lane = threadIdx.x & 31;
  int wv = threadIdx.x >> 5;  // 8 waves
  int l16 = lane & 15, hi = lane >> 4;
  int m0 = blockIdx.x * 16;
  const _Float16* arow = emb16 + (size_t)seq[m0 + l16] * 128;
  v16h a[4];
#pragma unroll
  for (int kk = 0; kk < 4; ++kk) a[kk] = load_a16(arow, kk * 32, hi);
#pragma unroll
  for (int i = 0; i < 4; ++i) {
    int n0 = (wv + 8 * i) * 16;
    v8f acc = {};
#pragma unroll
    for (int kk = 0; kk < 4; ++kk)
      acc = wmma_f16(a[kk], load_b16(wihx16, 128, n0, kk * 32, l16, hi), acc);
    int col = n0 + l16;
#pragma unroll
    for (int r = 0; r < 8; ++r) {
      int rr = m0 + hi * 8 + r;
      xw[(size_t)rr * 512 + col] = acc[r] + hbias[(rr >> 8) * 512 + col];
    }
  }
}

// ---------------------------------------------------------------------------
// Kernel 2: sequential LSTM scan, one workgroup (8 waves). Per step:
// gates[32,512] = h_prev(LDS) @ W_h.T (WMMA) + xw[:,t,:]; wave w owns the
// 16-wide j-band [16w,16w+16) for ALL four gates -> elementwise in registers.
// ---------------------------------------------------------------------------
__global__ void __launch_bounds__(256) scan_kernel(const _Float16* __restrict__ wh16,
                                                   const float* __restrict__ xw,
                                                   const float* __restrict__ c0g,
                                                   _Float16* __restrict__ hs16) {
  __shared__ __align__(16) _Float16 s_h[2][32][136];  // padded: conflict-free ds_b128
  __shared__ float s_c0[32][128];
  int tid = threadIdx.x;
  int lane = tid & 31, wv = tid >> 5;
  int l16 = lane & 15, hi = lane >> 4;
  for (int i = tid; i < 2 * 32 * 136; i += 256) ((_Float16*)s_h)[i] = (_Float16)0.0f;
  for (int i = tid; i < 32 * 128; i += 256) ((float*)s_c0)[i] = c0g[i];
  __syncthreads();

  int jb = wv * 16;  // this wave's hidden-index band
  int cur = 0;
  for (int t = 0; t < 256; ++t) {
    v16h a[2][4];
#pragma unroll
    for (int mt = 0; mt < 2; ++mt) {
      const _Float16* rowp = &s_h[cur][mt * 16 + l16][0];
#pragma unroll
      for (int kk = 0; kk < 4; ++kk) a[mt][kk] = load_a16(rowp, kk * 32, hi);
    }
    v8f acc[2][4];
#pragma unroll
    for (int mt = 0; mt < 2; ++mt)
#pragma unroll
      for (int g = 0; g < 4; ++g) {
        v8f c = {};
        int n0 = g * 128 + jb;
#pragma unroll
        for (int kk = 0; kk < 4; ++kk)
          c = wmma_f16(a[mt][kk], load_b16(wh16, 128, n0, kk * 32, l16, hi), c);
        acc[mt][g] = c;
      }
#pragma unroll
    for (int mt = 0; mt < 2; ++mt) {
#pragma unroll
      for (int r = 0; r < 8; ++r) {
        int b = mt * 16 + hi * 8 + r;
        size_t row = (size_t)b * 256 + t;
        const float* xp = xw + row * 512 + jb + l16;
        float gi = acc[mt][0][r] + xp[0];
        float gf = acc[mt][1][r] + xp[128];
        float gg = acc[mt][2][r] + xp[256];
        float go = acc[mt][3][r] + xp[384];
        float c = sigmoid_(gf) * s_c0[b][jb + l16] + sigmoid_(gi) * tanh_(gg);
        float h = sigmoid_(go) * tanh_(c);
        _Float16 h16 = (_Float16)h;
        s_h[cur ^ 1][b][jb + l16] = h16;
        hs16[row * 128 + jb + l16] = h16;
      }
    }
    __syncthreads();
    cur ^= 1;
  }
}

// ---------------------------------------------------------------------------
// Kernel 3: fused projection + log_softmax. Block = one 16-row tile of hs.
// Pass 1: WMMA over all V tiles, online per-row (max, sumexp) in registers;
// LDS tree-merge -> lse. Pass 2: recompute WMMA logits, write x+b-lse.
// W_pred16 (8 MB) is L2-resident; logits are never spilled to HBM.
// ---------------------------------------------------------------------------
__global__ void __launch_bounds__(256) proj_kernel(const _Float16* __restrict__ hs16,
                                                   const _Float16* __restrict__ wpred16,
                                                   const float* __restrict__ bpred,
                                                   float* __restrict__ out) {
  __shared__ float s_max[16][128];
  __shared__ float s_sum[16][128];
  __shared__ float s_lse[16];
  const int V = 32000;
  int tid = threadIdx.x, lane = tid & 31, wv = tid >> 5;
  int l16 = lane & 15, hi = lane >> 4;
  int m0 = blockIdx.x * 16;
  const _Float16* arow = hs16 + (size_t)(m0 + l16) * 128;
  v16h a[4];
#pragma unroll
  for (int kk = 0; kk < 4; ++kk) a[kk] = load_a16(arow, kk * 32, hi);

  float m8[8], s8[8];
#pragma unroll
  for (int r = 0; r < 8; ++r) { m8[r] = -1e30f; s8[r] = 0.0f; }

  int t0 = wv * 250, t1 = t0 + 250;  // 2000 N-tiles over 8 waves, contiguous bands
  for (int nt = t0; nt < t1; ++nt) {
    int n0 = nt * 16;
    v8f acc = {};
#pragma unroll
    for (int kk = 0; kk < 4; ++kk)
      acc = wmma_f16(a[kk], load_b16(wpred16, 128, n0, kk * 32, l16, hi), acc);
    float bp = bpred[n0 + l16];
#pragma unroll
    for (int r = 0; r < 8; ++r) {
      float x = acc[r] + bp;
      if (x > m8[r]) { s8[r] = s8[r] * __expf(m8[r] - x) + 1.0f; m8[r] = x; }
      else           { s8[r] += __expf(x - m8[r]); }
    }
  }
#pragma unroll
  for (int r = 0; r < 8; ++r) {
    s_max[hi * 8 + r][wv * 16 + l16] = m8[r];
    s_sum[hi * 8 + r][wv * 16 + l16] = s8[r];
  }
  __syncthreads();
  if (tid < 16) {
    float m = -1e30f, s = 0.0f;
    for (int i = 0; i < 128; ++i) {
      float pm = s_max[tid][i], ps = s_sum[tid][i];
      float mm = fmaxf(m, pm);
      s = s * __expf(m - mm) + ps * __expf(pm - mm);
      m = mm;
    }
    s_lse[tid] = m + __logf(s);
  }
  __syncthreads();
  for (int nt = t0; nt < t1; ++nt) {
    int n0 = nt * 16;
    v8f acc = {};
#pragma unroll
    for (int kk = 0; kk < 4; ++kk)
      acc = wmma_f16(a[kk], load_b16(wpred16, 128, n0, kk * 32, l16, hi), acc);
    float bp = bpred[n0 + l16];
#pragma unroll
    for (int r = 0; r < 8; ++r)
      out[(size_t)(m0 + hi * 8 + r) * V + n0 + l16] = acc[r] + bp - s_lse[hi * 8 + r];
  }
}

// ---------------------------------------------------------------------------
extern "C" void kernel_launch(void* const* d_in, const int* in_sizes, int n_in,
                              void* d_out, int out_size, void* d_ws, size_t ws_size,
                              hipStream_t stream) {
  (void)in_sizes; (void)n_in; (void)out_size; (void)ws_size;
  const int*   seq   = (const int*)  d_in[0];
  // d_in[1] = encoder_output: unused by the reference math
  const float* h0    = (const float*)d_in[2];   // [1,B,H]
  const float* c0    = (const float*)d_in[3];   // [1,B,H]
  const float* emb   = (const float*)d_in[4];   // [V,E]
  const float* wih   = (const float*)d_in[5];   // [4H, E+H]
  const float* bih   = (const float*)d_in[6];   // [4H]
  const float* whh   = (const float*)d_in[7];   // [4H, H]
  const float* bhh   = (const float*)d_in[8];   // [4H]
  const float* wpred = (const float*)d_in[9];   // [V, H]
  const float* bpred = (const float*)d_in[10];  // [V]
  float* out = (float*)d_out;

  char* ws = (char*)d_ws;
  size_t off = 0;
  auto alloc = [&](size_t bytes) -> void* {
    void* p = ws + off;
    off = (off + bytes + 255) & ~(size_t)255;
    return p;
  };
  _Float16* emb16   = (_Float16*)alloc((size_t)32000 * 128 * 2);
  _Float16* wpred16 = (_Float16*)alloc((size_t)32000 * 128 * 2);
  _Float16* wihx16  = (_Float16*)alloc((size_t)512 * 128 * 2);
  _Float16* wh16    = (_Float16*)alloc((size_t)512 * 128 * 2);
  float*    hbias   = (float*)   alloc((size_t)32 * 512 * 4);
  float*    xw      = (float*)   alloc((size_t)8192 * 512 * 4);
  _Float16* hs16    = (_Float16*)alloc((size_t)8192 * 128 * 2);

  hipLaunchKernelGGL(prep_kernel, dim3(2048), dim3(256), 0, stream,
                     emb, wih, whh, bih, bhh, h0, wpred,
                     emb16, wpred16, wihx16, wh16, hbias);
  hipLaunchKernelGGL(xw_kernel, dim3(512), dim3(256), 0, stream,
                     seq, emb16, wihx16, hbias, xw);
  hipLaunchKernelGGL(scan_kernel, dim3(1), dim3(256), 0, stream,
                     wh16, xw, c0, hs16);
  hipLaunchKernelGGL(proj_kernel, dim3(512), dim3(256), 0, stream,
                     hs16, wpred16, bpred, out);
}
